// GlmAsrEncoderAttention_86303072846571
// MI455X (gfx1250) — matrix operand order repaired and checked by reading, not compiled, not run on hardware
//
#include <hip/hip_runtime.h>
#include <hip/hip_bf16.h>

typedef __bf16 bf16;
typedef __attribute__((ext_vector_type(16))) __bf16 v16bf;
typedef __attribute__((ext_vector_type(8)))  __bf16 v8bf;
typedef __attribute__((ext_vector_type(8)))  float  v8f;

#define B_    4
#define S_    2048
#define HID_  2048
#define NH_   16
#define NKV_  4
#define HD_   128
#define MTOT  (B_ * S_)          // 8192 token rows
#define NQKV  3072               // Q(2048) + K(512) + V(512)

// ---------------------------------------------------------------------------
// Fragment helpers (CDNA5 wave32 WMMA layouts, CDNA5 ISA §7.12.2)
// A (16x32 bf16): lane -> row (lane%16); element e -> k = (e/8)*16 + (lane/16)*8 + e%8
// B (32x16 bf16): lane -> col (lane%16); element e -> k, same k order as A.
// C/D (16x16 f32): element r -> row r + 8*(lane/16); col = lane%16.
// ---------------------------------------------------------------------------
static __device__ __forceinline__ v16bf load_afrag(const bf16* p, int hf) {
  const v8bf lo = *(const v8bf*)(p + hf * 8);
  const v8bf hi = *(const v8bf*)(p + 16 + hf * 8);
  return __builtin_shufflevector(lo, hi, 0,1,2,3,4,5,6,7,8,9,10,11,12,13,14,15);
}

static __device__ __forceinline__ v8f wmma_bf16(v16bf a, v16bf b, v8f c) {
  return __builtin_amdgcn_wmma_f32_16x16x32_bf16(false, a, false, b, (short)0, c,
                                                 false, false);
}

// CDNA5 async tile staging: memory -> LDS without touching VGPR data path.
// Inline asm is portable across ROCm-7.2 / clang-23 toolchains (builtin arity
// differs between them). Tracked by ASYNCcnt (CDNA5 ISA).
static __device__ __forceinline__ void async_b128(const bf16* g, bf16* l) {
  asm volatile("global_load_async_to_lds_b128 %0, %1, off"
               :: "v"((unsigned)(unsigned long long)(uintptr_t)l),
                  "v"((unsigned long long)(uintptr_t)g)
               : "memory");
}
static __device__ __forceinline__ void wait_async0() {
  asm volatile("s_wait_asynccnt 0" ::: "memory");
}

// ---------------------------------------------------------------------------
// fp32 -> bf16 conversion
// ---------------------------------------------------------------------------
__global__ void k_f32_to_bf16(const float* __restrict__ src,
                              bf16* __restrict__ dst, int n) {
  int i = blockIdx.x * blockDim.x + threadIdx.x;
  if (i < n) dst[i] = (bf16)src[i];
}

// ---------------------------------------------------------------------------
// LDS-tiled bf16 GEMM: C[M, NCOLS] = A[M, 2048] * B[2048, NCOLS] + bias
// Block tile 128x64, K-step 32, 8 waves as 4(M) x 2(N), each wave 32x32.
// Double-buffered LDS fed by GLOBAL_LOAD_ASYNC_TO_LDS_B128.
// ---------------------------------------------------------------------------
template <int NCOLS, bool WRITE_F32>
__global__ __launch_bounds__(256) void k_gemm_bf16(const bf16* __restrict__ A,
                                                   const bf16* __restrict__ Bm,
                                                   const float* __restrict__ bias,
                                                   void* __restrict__ out) {
  __shared__ bf16 lds_a[2][128 * 32];
  __shared__ bf16 lds_b[2][32 * 64];
  const int tid  = threadIdx.x;
  const int lane = tid & 31;
  const int w    = tid >> 5;
  const int wm   = w & 3, wn = w >> 2;
  const int hf   = lane >> 4;
  const int lrow = lane & 15;
  const int gm0  = blockIdx.y * 128;
  const int gn0  = blockIdx.x * 64;

  // cooperative staging coords: 16B (8 bf16) chunks
  const int ar = tid >> 2, as = tid & 3;   // A: rows ar and ar+64, segment as
  const int br = tid >> 3, bs = tid & 7;   // B: row br, segment bs

  auto stage = [&](int kb, int buf) {
    async_b128(&A[(size_t)(gm0 + ar) * HID_ + kb + as * 8],
               &lds_a[buf][ar * 32 + as * 8]);
    async_b128(&A[(size_t)(gm0 + ar + 64) * HID_ + kb + as * 8],
               &lds_a[buf][(ar + 64) * 32 + as * 8]);
    async_b128(&Bm[(size_t)(kb + br) * NCOLS + gn0 + bs * 8],
               &lds_b[buf][br * 64 + bs * 8]);
  };

  stage(0, 0);
  wait_async0();
  __syncthreads();

  v8f acc[2][2] = {};
  for (int it = 0; it < HID_ / 32; ++it) {
    const int buf = it & 1;
    if (it + 1 < HID_ / 32) stage((it + 1) * 32, buf ^ 1);

    v16bf af[2], bfr[2];
#pragma unroll
    for (int i = 0; i < 2; ++i)
      af[i] = load_afrag(&lds_a[buf][(wm * 32 + i * 16 + lrow) * 32], hf);
#pragma unroll
    for (int j = 0; j < 2; ++j)
#pragma unroll
      for (int e = 0; e < 16; ++e)
        bfr[j][e] = lds_b[buf][((e >> 3) * 16 + hf * 8 + (e & 7)) * 64 + wn * 32 + j * 16 + lrow];
#pragma unroll
    for (int i = 0; i < 2; ++i)
#pragma unroll
      for (int j = 0; j < 2; ++j)
        acc[i][j] = wmma_bf16(af[i], bfr[j], acc[i][j]);

    wait_async0();   // next tile resident in LDS
    __syncthreads(); // everyone done reading buf before it is refilled
  }

#pragma unroll
  for (int i = 0; i < 2; ++i) {
#pragma unroll
    for (int j = 0; j < 2; ++j) {
      const int col = gn0 + wn * 32 + j * 16 + lrow;
      const float bv = bias[col];
#pragma unroll
      for (int r = 0; r < 8; ++r) {
        const int row = gm0 + wm * 32 + i * 16 + r + 8 * hf;
        const float v = acc[i][j][r] + bv;
        if (WRITE_F32)
          ((float*)out)[(size_t)row * NCOLS + col] = v;
        else
          ((bf16*)out)[(size_t)row * NCOLS + col] = (bf16)v;
      }
    }
  }
}

// ---------------------------------------------------------------------------
// RoPE + re-layout kernels
// ---------------------------------------------------------------------------
__global__ void k_rope_q(const bf16* __restrict__ qkv, const float* __restrict__ cosT,
                         const float* __restrict__ sinT, bf16* __restrict__ qout) {
  int idx = blockIdx.x * blockDim.x + threadIdx.x;  // B*S*NH*64
  int d = idx & 63; int t = idx >> 6;
  int h = t & 15;   t >>= 4;
  int s = t & 2047; int b = t >> 11;
  size_t row = (size_t)(b * S_ + s) * NQKV + h * HD_;
  float x1 = (float)qkv[row + d];
  float x2 = (float)qkv[row + d + 64];
  float c = cosT[s * 64 + d], sn = sinT[s * 64 + d];
  size_t o = ((size_t)(b * NH_ + h) * S_ + s) * HD_;
  qout[o + d]      = (bf16)(x1 * c - x2 * sn);
  qout[o + d + 64] = (bf16)(x2 * c + x1 * sn);
}

__global__ void k_rope_k(const bf16* __restrict__ qkv, const float* __restrict__ cosT,
                         const float* __restrict__ sinT, bf16* __restrict__ ktout) {
  int idx = blockIdx.x * blockDim.x + threadIdx.x;  // B*S*NKV*64
  int d = idx & 63; int t = idx >> 6;
  int kh = t & 3;   t >>= 2;
  int s = t & 2047; int b = t >> 11;
  size_t row = (size_t)(b * S_ + s) * NQKV + 2048 + kh * HD_;
  float x1 = (float)qkv[row + d];
  float x2 = (float)qkv[row + d + 64];
  float c = cosT[s * 64 + d], sn = sinT[s * 64 + d];
  // K stored as [B, NKV, S, HD] so K^T B-fragments (k = d) are contiguous per key
  size_t ob = ((size_t)(b * NKV_ + kh) * S_ + s) * HD_;
  ktout[ob + d]      = (bf16)(x1 * c - x2 * sn);
  ktout[ob + d + 64] = (bf16)(x2 * c + x1 * sn);
}

__global__ void k_copy_v(const bf16* __restrict__ qkv, bf16* __restrict__ vout) {
  int idx = blockIdx.x * blockDim.x + threadIdx.x;  // B*S*NKV*128
  int d = idx & 127; int t = idx >> 7;
  int kh = t & 3;    t >>= 2;
  int s = t & 2047;  int b = t >> 11;
  vout[((size_t)(b * NKV_ + kh) * S_ + s) * HD_ + d] =
      qkv[(size_t)(b * S_ + s) * NQKV + 2560 + kh * HD_ + d];
}

// ---------------------------------------------------------------------------
// Flash attention (full, non-causal). One block per (q-block of 128, head, batch).
// Wave w owns 16 q-rows. Key blocks of 64. Online softmax in f32.
// K/V fragments stream straight from L2 (192 MB; per-head K/V is ~1 MB) and the
// compiler pipelines the clause loads against WMMA with partial s_wait_loadcnt.
// ---------------------------------------------------------------------------
__global__ __launch_bounds__(256) void k_attn(const bf16* __restrict__ Q,
                                              const bf16* __restrict__ Kt,
                                              const bf16* __restrict__ V,
                                              bf16* __restrict__ Oatt) {
  __shared__ bf16 lds_p[8 * 16 * 64];  // per-wave P staging (D-layout -> A-layout)
  const int tid = threadIdx.x, lane = tid & 31, w = tid >> 5;
  const int hf = lane >> 4, lrow = lane & 15;
  const int qblk = blockIdx.x, h = blockIdx.y, b = blockIdx.z;
  const int kvh = h >> 2;  // GQA: 4 query heads per KV head

  const bf16* Qb  = Q  + (size_t)(b * NH_ + h)  * S_  * HD_;
  const bf16* Ktb = Kt + (size_t)(b * NKV_ + kvh) * HD_ * S_;
  const bf16* Vb  = V  + (size_t)(b * NKV_ + kvh) * S_  * HD_;
  const int q0 = qblk * 128 + w * 16;

  v16bf qf[4];
#pragma unroll
  for (int t = 0; t < 4; ++t)
    qf[t] = load_afrag(Qb + (size_t)(q0 + lrow) * HD_ + t * 32, hf);

  v8f o[8] = {};
  float mrun[8], lrun[8];
#pragma unroll
  for (int r = 0; r < 8; ++r) { mrun[r] = -1e30f; lrun[r] = 0.0f; }
  const float scale = 0.08838834764831845f;  // 1/sqrt(128)
  bf16* pl = &lds_p[w * 16 * 64];

  for (int n0 = 0; n0 < S_; n0 += 64) {
    // S = Q @ K^T for 16x64 tile
    v8f sfr[4] = {};
#pragma unroll
    for (int j = 0; j < 4; ++j)
#pragma unroll
      for (int t = 0; t < 4; ++t) {
        v16bf kb = load_afrag(Ktb + (size_t)(n0 + j * 16 + lrow) * HD_ + t * 32, hf);
        sfr[j] = wmma_bf16(qf[t], kb, sfr[j]);
      }

    // online softmax (row stats replicated across the 16 lanes of each half)
#pragma unroll
    for (int r = 0; r < 8; ++r) {
      float mx = -1e30f;
#pragma unroll
      for (int j = 0; j < 4; ++j) { sfr[j][r] *= scale; mx = fmaxf(mx, sfr[j][r]); }
#pragma unroll
      for (int off = 1; off < 16; off <<= 1) mx = fmaxf(mx, __shfl_xor(mx, off, 32));
      const float nm = fmaxf(mrun[r], mx);
      const float fsc = __expf(mrun[r] - nm);
      float rs = 0.0f;
#pragma unroll
      for (int j = 0; j < 4; ++j) {
        const float p = __expf(sfr[j][r] - nm);
        sfr[j][r] = p;
        rs += p;
      }
#pragma unroll
      for (int off = 1; off < 16; off <<= 1) rs += __shfl_xor(rs, off, 32);
      lrun[r] = lrun[r] * fsc + rs;
      mrun[r] = nm;
#pragma unroll
      for (int t = 0; t < 8; ++t) o[t][r] *= fsc;
    }

    // transpose P through per-wave LDS region (LDS is in-order per wave)
#pragma unroll
    for (int r = 0; r < 8; ++r) {
      const int prow = r + 8 * hf;
#pragma unroll
      for (int j = 0; j < 4; ++j) pl[prow * 64 + j * 16 + lrow] = (bf16)sfr[j][r];
    }

    // O += P @ V
#pragma unroll
    for (int pi = 0; pi < 2; ++pi) {
      v16bf pa = load_afrag(&pl[lrow * 64 + pi * 32], hf);
#pragma unroll
      for (int t2 = 0; t2 < 8; ++t2) {
        v16bf bv;
#pragma unroll
        for (int e = 0; e < 16; ++e)
          bv[e] = Vb[(size_t)(n0 + pi * 32 + (e >> 3) * 16 + hf * 8 + (e & 7)) * HD_ + t2 * 16 + lrow];
        o[t2] = wmma_bf16(pa, bv, o[t2]);
      }
    }
  }

  // normalize and write [B*S, HID] bf16 for the O-projection GEMM
#pragma unroll
  for (int t2 = 0; t2 < 8; ++t2)
#pragma unroll
    for (int r = 0; r < 8; ++r) {
      const int qr = q0 + r + 8 * hf;
      const float val = o[t2][r] / lrun[r];
      Oatt[(size_t)(b * S_ + qr) * HID_ + h * HD_ + t2 * 16 + lrow] = (bf16)val;
    }
}

// ---------------------------------------------------------------------------
// Host-side orchestration
// ---------------------------------------------------------------------------
extern "C" void kernel_launch(void* const* d_in, const int* in_sizes, int n_in,
                              void* d_out, int out_size, void* d_ws, size_t ws_size,
                              hipStream_t stream) {
  (void)in_sizes; (void)n_in; (void)out_size; (void)ws_size;
  const float* hid  = (const float*)d_in[0];
  const float* cosT = (const float*)d_in[1];
  const float* sinT = (const float*)d_in[2];
  const float* wqkv = (const float*)d_in[3];
  const float* bqkv = (const float*)d_in[4];
  const float* wo   = (const float*)d_in[5];
  const float* bo   = (const float*)d_in[6];

  char* ws = (char*)d_ws;
  // workspace layout (bytes); attn output aliases hid_bf (dead after GEMM1)
  bf16* hid_bf  = (bf16*)(ws);                          // 32 MB
  bf16* wqkv_bf = (bf16*)(ws + 33554432ull);            // 12 MB
  bf16* wo_bf   = (bf16*)(ws + 46137344ull);            // 8 MB
  bf16* qkv_bf  = (bf16*)(ws + 54525952ull);            // 48 MB
  bf16* q_buf   = (bf16*)(ws + 104857600ull);           // 32 MB
  bf16* kt_buf  = (bf16*)(ws + 138412032ull);           // 8 MB
  bf16* v_buf   = (bf16*)(ws + 146800640ull);           // 8 MB
  bf16* attn_bf = hid_bf;                               // alias

  const int nh = MTOT * HID_;          // 16,777,216
  const int nw = HID_ * NQKV;          // 6,291,456
  const int no = HID_ * HID_;          // 4,194,304
  k_f32_to_bf16<<<(nh + 255) / 256, 256, 0, stream>>>(hid, hid_bf, nh);
  k_f32_to_bf16<<<(nw + 255) / 256, 256, 0, stream>>>(wqkv, wqkv_bf, nw);
  k_f32_to_bf16<<<(no + 255) / 256, 256, 0, stream>>>(wo, wo_bf, no);

  // QKV projection + bias
  k_gemm_bf16<NQKV, false><<<dim3(NQKV / 64, MTOT / 128), 256, 0, stream>>>(
      hid_bf, wqkv_bf, bqkv, (void*)qkv_bf);

  // RoPE + re-layout
  k_rope_q<<<(B_ * S_ * NH_ * 64) / 256, 256, 0, stream>>>(qkv_bf, cosT, sinT, q_buf);
  k_rope_k<<<(B_ * S_ * NKV_ * 64) / 256, 256, 0, stream>>>(qkv_bf, cosT, sinT, kt_buf);
  k_copy_v<<<(B_ * S_ * NKV_ * HD_) / 256, 256, 0, stream>>>(qkv_bf, v_buf);

  // flash attention
  k_attn<<<dim3(S_ / 128, NH_, B_), 256, 0, stream>>>(q_buf, kt_buf, v_buf, attn_bf);

  // output projection + bias -> f32 d_out
  k_gemm_bf16<HID_, true><<<dim3(HID_ / 64, MTOT / 128), 256, 0, stream>>>(
      attn_bf, wo_bf, bo, d_out);
}
